// RNNModel_27084063768645
// MI455X (gfx1250) — matrix operand order repaired and checked by reading, not compiled, not run on hardware
//
#include <hip/hip_runtime.h>
#include <hip/hip_bf16.h>

typedef __attribute__((ext_vector_type(16))) _Float16 v16h;
typedef __attribute__((ext_vector_type(8)))  _Float16 half8;
typedef __attribute__((ext_vector_type(8)))  float    v8f;

#define TB   64      // batch
#define TT   1024    // time steps
#define DD   64      // input dim
#define HH   256     // hidden dim
#define OO   10      // output dim
#define NWG  16      // 8 WGs per layer
#define NTHR 256     // 8 wave32 per WG

// ---------------------------------------------------------------------------
// Prep: f32 -> f16 conversion of x and weights, zero state, reset barrier.
// Re-executed on every launch (graph replay safe / deterministic).
// ---------------------------------------------------------------------------
__global__ void prep_kernel(const float* __restrict__ x,
                            const float* __restrict__ wih0, const float* __restrict__ whh0,
                            const float* __restrict__ wih1, const float* __restrict__ whh1,
                            _Float16* __restrict__ x16,
                            _Float16* __restrict__ wih0h, _Float16* __restrict__ whh0h,
                            _Float16* __restrict__ wih1h, _Float16* __restrict__ whh1h,
                            _Float16* __restrict__ h0buf, _Float16* __restrict__ h1buf,
                            float* __restrict__ hsum, unsigned* __restrict__ bar) {
  long i0 = (long)blockIdx.x * blockDim.x + threadIdx.x;
  long st = (long)gridDim.x * blockDim.x;
  for (long i = i0; i < (long)TB * TT * DD; i += st) x16[i]   = (_Float16)x[i];
  for (long i = i0; i < 4L * HH * DD;       i += st) wih0h[i] = (_Float16)wih0[i];
  for (long i = i0; i < 4L * HH * HH;       i += st) whh0h[i] = (_Float16)whh0[i];
  for (long i = i0; i < 4L * HH * HH;       i += st) wih1h[i] = (_Float16)wih1[i];
  for (long i = i0; i < 4L * HH * HH;       i += st) whh1h[i] = (_Float16)whh1[i];
  for (long i = i0; i < 2L * TB * HH;       i += st) {
    h0buf[i] = (_Float16)0.0f;
    h1buf[i] = (_Float16)0.0f;
  }
  for (long i = i0; i < (long)TB * HH;      i += st) hsum[i] = 0.0f;
  if (i0 < 2) bar[i0] = 0u;
}

// ---------------------------------------------------------------------------
// Grid-wide barrier over NWG resident workgroups (atomic count + generation).
// ---------------------------------------------------------------------------
__device__ __forceinline__ void grid_barrier(unsigned* cnt, unsigned* gen, unsigned n) {
  __threadfence();      // make this thread's global stores visible device-wide
  __syncthreads();
  if (threadIdx.x == 0) {
    volatile unsigned* vgen = (volatile unsigned*)gen;
    unsigned g = *vgen;
    if (atomicAdd(cnt, 1u) == n - 1u) {
      *(volatile unsigned*)cnt = 0u;
      __threadfence();
      *vgen = g + 1u;
    } else {
      while (*vgen == g) { __builtin_amdgcn_s_sleep(2); }
    }
  }
  __syncthreads();
  __threadfence();      // acquire side for all threads
}

// ---------------------------------------------------------------------------
// One GEMM phase: acc[mt (4 m-tiles)] += A[16mt.., k] * W[col,k]^T over K.
// A: row-major [M, lda] f16.  Wrow: this lane's weight row (= B column), K contiguous.
// CDNA5 v16h A layout: lane half h, VGPR v<4 -> k0+8h+2v..+1 ; v>=4 -> k0+16+8h+...
//   => two contiguous 16B (half8) loads at +8h and +16+8h.
// CDNA5 v16h B layout: same k order as A: VGPR v<4 -> k0+8h+2v..+1 ; v>=4 -> k0+16+8h+...
//   => two contiguous 16B (half8) loads at +8h and +16+8h along the W row.
// ---------------------------------------------------------------------------
__device__ __forceinline__ void wmma_phase(const _Float16* __restrict__ A, int lda,
                                           const _Float16* __restrict__ Wrow, int K,
                                           int lane16, int half, v8f acc[4]) {
  for (int k0 = 0; k0 < K; k0 += 32) {
    const _Float16* bp = Wrow + k0 + 8 * half;
    half8 b_lo = *(const half8*)(bp);
    half8 b_hi = *(const half8*)(bp + 16);
    v16h bf;
#pragma unroll
    for (int i = 0; i < 8; ++i) { bf[i] = b_lo[i]; bf[8 + i] = b_hi[i]; }
#pragma unroll
    for (int mt = 0; mt < 4; ++mt) {
      const _Float16* ap = A + (size_t)(16 * mt + lane16) * lda + k0;
      half8 a_lo = *(const half8*)(ap + 8 * half);
      half8 a_hi = *(const half8*)(ap + 16 + 8 * half);
      v16h af;
#pragma unroll
      for (int i = 0; i < 8; ++i) { af[i] = a_lo[i]; af[8 + i] = a_hi[i]; }
      acc[mt] = __builtin_amdgcn_wmma_f32_16x16x32_f16(
          /*neg_a=*/false, af, /*neg_b=*/false, bf,
          /*c_mod=*/(short)0, acc[mt], /*reuse_a=*/false, /*reuse_b=*/false);
    }
  }
}

// ---------------------------------------------------------------------------
// Persistent pipelined 2-layer LSTM.
// WGs 0..7  : layer 0, hidden slice [32s, 32s+32), active iters k = 0..T-1
// WGs 8..15 : layer 1, one step behind,             active iters k = 1..T
// Per iter: gates[64 x 128] = x_in @ Wih_slice^T + h_prev @ Whh_slice^T (WMMA),
// stage in LDS, elementwise LSTM update (c in regs), h -> f16 ping-pong buffer,
// grid barrier.
// ---------------------------------------------------------------------------
__global__ void __launch_bounds__(NTHR) lstm_kernel(
    const _Float16* __restrict__ x16,
    const _Float16* __restrict__ wih0, const _Float16* __restrict__ whh0,
    const _Float16* __restrict__ wih1, const _Float16* __restrict__ whh1,
    const float* __restrict__ bih0, const float* __restrict__ bhh0,
    const float* __restrict__ bih1, const float* __restrict__ bhh1,
    _Float16* __restrict__ h0buf, _Float16* __restrict__ h1buf,
    float* __restrict__ hsum, unsigned* __restrict__ bar) {
  __shared__ float gates[TB * 128];  // 32 KB gate staging
  __shared__ float bsum[128];        // combined bias slice (col-local: gate*32 + j)

  const int wg     = blockIdx.x;
  const int layer  = wg >> 3;
  const int s      = wg & 7;          // hidden-slice id: units [32s, 32s+32)
  const int tid    = threadIdx.x;
  const int w      = tid >> 5;        // wave id 0..7 -> owns n-tile w (cols 16w..16w+15)
  const int lane   = tid & 31;
  const int lane16 = lane & 15;
  const int half   = lane >> 4;

  const _Float16* Wih = layer ? wih1 : wih0;
  const _Float16* Whh = layer ? whh1 : whh0;
  const float*    bih = layer ? bih1 : bih0;
  const float*    bhh = layer ? bhh1 : bhh0;
  const int       Kx  = layer ? HH : DD;   // K of the x-input GEMM

  if (tid < 128) {  // combined bias for this WG's 128 gate columns
    int gate = tid >> 5, j = tid & 31;
    int gi = gate * HH + s * 32 + j;
    bsum[tid] = bih[gi] + bhh[gi];
  }

  // weight row (== B-matrix column) owned by this lane within wave w's n-tile
  const int colg = (w >> 1) * HH + s * 32 + (w & 1) * 16 + lane16;
  const _Float16* WihRow = Wih + (size_t)colg * Kx;
  const _Float16* WhhRow = Whh + (size_t)colg * HH;

  // elementwise ownership: batch eb, hidden j = ejb..ejb+7 (local)
  const int eb  = tid >> 2;
  const int ejb = (tid & 3) * 8;
  float cst[8], hs[8];
#pragma unroll
  for (int r = 0; r < 8; ++r) { cst[r] = 0.0f; hs[r] = 0.0f; }

  __syncthreads();

  for (int k = 0; k <= TT; ++k) {
    const bool active = (layer == 0) ? (k < TT) : (k >= 1);
    if (active) {
      const int t = (layer == 0) ? k : (k - 1);
      _Float16* mybuf = layer ? h1buf : h0buf;
      const _Float16* hprev = mybuf + (size_t)(k & 1) * TB * HH;        // h[t-1]
      _Float16*       hnext = mybuf + (size_t)((k + 1) & 1) * TB * HH;  // h[t]
      // x-input: layer0 reads x16[b][t][:], layer1 reads h0[t] (= h0buf[k&1])
      const _Float16* Ax  = (layer == 0) ? (x16 + (size_t)t * DD)
                                         : (h0buf + (size_t)(k & 1) * TB * HH);
      const int       ldx = (layer == 0) ? (TT * DD) : HH;

      v8f acc[4];
#pragma unroll
      for (int mt = 0; mt < 4; ++mt)
#pragma unroll
        for (int i = 0; i < 8; ++i) acc[mt][i] = 0.0f;

      wmma_phase(Ax,    ldx, WihRow, Kx, lane16, half, acc);  // x @ Wih^T
      wmma_phase(hprev, HH,  WhhRow, HH, lane16, half, acc);  // h @ Whh^T

      // C/D layout: VGPR v holds m_local = v + 8*half, n = lane16; tile n-base = 16w
#pragma unroll
      for (int mt = 0; mt < 4; ++mt)
#pragma unroll
        for (int v = 0; v < 8; ++v)
          gates[(16 * mt + v + 8 * half) * 128 + 16 * w + lane16] = acc[mt][v];
      __syncthreads();

#pragma unroll
      for (int r = 0; r < 8; ++r) {
        int j = ejb + r;
        float gi = gates[eb * 128 +       j] + bsum[j];
        float gf = gates[eb * 128 +  32 + j] + bsum[32 + j];
        float gg = gates[eb * 128 +  64 + j] + bsum[64 + j];
        float go = gates[eb * 128 +  96 + j] + bsum[96 + j];
        float iv = 1.0f / (1.0f + __expf(-gi));
        float fv = 1.0f / (1.0f + __expf(-gf));
        float gv = tanhf(gg);
        float ov = 1.0f / (1.0f + __expf(-go));
        cst[r] = fv * cst[r] + iv * gv;
        float hv = ov * tanhf(cst[r]);
        hnext[eb * HH + s * 32 + j] = (_Float16)hv;
        if (layer) hs[r] += hv;      // running sum for the time-mean
      }
    }
    grid_barrier(bar, bar + 1, NWG);  // includes __syncthreads (covers LDS reuse)
  }

  if (layer) {
#pragma unroll
    for (int r = 0; r < 8; ++r)
      hsum[eb * HH + s * 32 + ejb + r] = hs[r];
  }
}

// ---------------------------------------------------------------------------
// out[b][o] = b_fc[o] + (1/T) * dot(hsum[b], W_fc[o])   (640 outputs, tiny)
// ---------------------------------------------------------------------------
__global__ void fc_kernel(const float* __restrict__ hsum,
                          const float* __restrict__ wfc,
                          const float* __restrict__ bfc,
                          float* __restrict__ out) {
  int idx = blockIdx.x * blockDim.x + threadIdx.x;
  if (idx >= TB * OO) return;
  int b = idx / OO, o = idx % OO;
  float acc = 0.0f;
  for (int kk = 0; kk < HH; ++kk) acc += hsum[b * HH + kk] * wfc[o * HH + kk];
  out[idx] = acc * (1.0f / (float)TT) + bfc[o];
}

// ---------------------------------------------------------------------------
extern "C" void kernel_launch(void* const* d_in, const int* in_sizes, int n_in,
                              void* d_out, int out_size, void* d_ws, size_t ws_size,
                              hipStream_t stream) {
  const float* x    = (const float*)d_in[0];
  const float* wih0 = (const float*)d_in[1];
  const float* whh0 = (const float*)d_in[2];
  const float* bih0 = (const float*)d_in[3];
  const float* bhh0 = (const float*)d_in[4];
  const float* wih1 = (const float*)d_in[5];
  const float* whh1 = (const float*)d_in[6];
  const float* bih1 = (const float*)d_in[7];
  const float* bhh1 = (const float*)d_in[8];
  const float* wfc  = (const float*)d_in[9];
  const float* bfc  = (const float*)d_in[10];

  char* ws = (char*)d_ws;
  // workspace layout (all offsets 256B-aligned); total ~10.3 MB
  _Float16* x16   = (_Float16*)(ws + 0);                 // 8,388,608 B
  _Float16* wih0h = (_Float16*)(ws + 8388608);           //   131,072 B
  _Float16* whh0h = (_Float16*)(ws + 8519680);           //   524,288 B
  _Float16* wih1h = (_Float16*)(ws + 9043968);           //   524,288 B
  _Float16* whh1h = (_Float16*)(ws + 9568256);           //   524,288 B
  _Float16* h0buf = (_Float16*)(ws + 10092544);          //    65,536 B (2x ping-pong)
  _Float16* h1buf = (_Float16*)(ws + 10158080);          //    65,536 B
  float*    hsum  = (float*)   (ws + 10223616);          //    65,536 B
  unsigned* bar   = (unsigned*)(ws + 10289152);          //         8 B

  prep_kernel<<<1024, 256, 0, stream>>>(x, wih0, whh0, wih1, whh1,
                                        x16, wih0h, whh0h, wih1h, whh1h,
                                        h0buf, h1buf, hsum, bar);

  lstm_kernel<<<NWG, NTHR, 0, stream>>>(x16, wih0h, whh0h, wih1h, whh1h,
                                        bih0, bhh0, bih1, bhh1,
                                        h0buf, h1buf, hsum, bar);

  fc_kernel<<<(TB * OO + 255) / 256, 256, 0, stream>>>(hsum, wfc, bfc, (float*)d_out);
}
